// LinearAttention_63479616635170
// MI455X (gfx1250) — matrix operand order repaired and checked by reading, not compiled, not run on hardware
//
#include <hip/hip_runtime.h>
#include <cmath>

typedef __attribute__((ext_vector_type(16))) _Float16 v16h;
typedef __attribute__((ext_vector_type(8)))  float    v8f;

#define B_SZ     8
#define DIM      128
#define NTOK     32768
#define NHEADS   4
#define DHEAD    32
#define QKV_ROWS 384
#define SCALE_F  0.17677669529663687f
#define EPS_F    1e-5f

__device__ __forceinline__ v8f wmma_f16(v16h a, v16h b, v8f c) {
  // D = A(16x32 f16) * B(32x16 f16) + C(16x16 f32)
  return __builtin_amdgcn_wmma_f32_16x16x32_f16(
      /*neg_a=*/false, a, /*neg_b=*/false, b,
      /*c_mod=*/(short)0, c, /*reuse_a=*/false, /*reuse_b=*/false);
}

// A fragment (16x32 f16, M=lane row): lanes 0-15 K in [kk, kk+8)+[kk+16, kk+24),
// lanes 16-31 K in [kk+8, kk+16)+[kk+24, kk+32). src = row-major f32, ld = row stride.
__device__ __forceinline__ v16h load_a_frag(const float* __restrict__ src) {
  v16h a;
  #pragma unroll
  for (int j = 0; j < 8; ++j) a[j]     = (_Float16)src[j];
  #pragma unroll
  for (int j = 0; j < 8; ++j) a[8 + j] = (_Float16)src[16 + j];
  return a;
}

// ---------------------------------------------------------------------------
// K1: qkv[b, o, n] = sum_c w_qkv[o,c] * x[b,c,n]   (M=384, K=128, N=32768)
// One wave = one 16-wide n-tile x SIX 16-row m-tiles: x fragment built once
// per K-step and reused for 6 WMMAs (x read amplification 24x -> 4x).
// ---------------------------------------------------------------------------
__global__ __launch_bounds__(256) void qkv_gemm_kernel(
    const float* __restrict__ x, const float* __restrict__ w_qkv,
    float* __restrict__ qkv)
{
  const int lane = threadIdx.x & 31;
  const int wave = threadIdx.x >> 5;
  const int half = lane >> 4;     // lane group (0: lanes 0-15, 1: 16-31)
  const int lrow = lane & 15;
  const int b     = blockIdx.z;
  const int mbase = blockIdx.y * 96;                 // 6 tiles of 16 rows
  const int n0    = (blockIdx.x * 8 + wave) * 16;

  const size_t xb = (size_t)b * DIM * NTOK;
  v8f acc[6] = {};
  #pragma unroll
  for (int kk = 0; kk < DIM; kk += 32) {
    // B fragment (32x16): lane = column n, elem j -> K = kk + half*8 + (j < 8 ? j : j + 8)
    v16h bf;
    const float* xcol = x + xb + (size_t)(kk + half * 8) * NTOK + n0 + lrow;
    #pragma unroll
    for (int j = 0; j < 16; ++j) bf[j] = (_Float16)xcol[(size_t)(j < 8 ? j : j + 8) * NTOK];
    #pragma unroll
    for (int t = 0; t < 6; ++t) {
      const v16h a = load_a_frag(
          w_qkv + (size_t)(mbase + t * 16 + lrow) * DIM + kk + half * 8);
      acc[t] = wmma_f16(a, bf, acc[t]);
    }
  }
  float* qb = qkv + (size_t)b * QKV_ROWS * NTOK;
  #pragma unroll
  for (int t = 0; t < 6; ++t)
    #pragma unroll
    for (int r = 0; r < 8; ++r)   // D: VGPR r -> M = half*8 + r, N = lrow
      qb[(size_t)(mbase + t * 16 + half * 8 + r) * NTOK + n0 + lrow] = acc[t][r];
}

// ---------------------------------------------------------------------------
// K2: per-row (b,h,d) max and sum(exp) over n for the k softmax
// ---------------------------------------------------------------------------
__global__ __launch_bounds__(256) void k_stats_kernel(
    const float* __restrict__ qkv, float* __restrict__ rowmax,
    float* __restrict__ rowsum)
{
  __shared__ float sm[256];
  const int rid = blockIdx.x;            // 0..1023 = b*128 + h*32 + d
  const int b   = rid >> 7;
  const int hd  = rid & 127;
  const float* row = qkv + ((size_t)b * QKV_ROWS + DIM + hd) * NTOK;
  const int tid = threadIdx.x;
  float m = -3.4e38f;
  for (int i = tid; i < NTOK; i += 256) m = fmaxf(m, row[i]);
  sm[tid] = m; __syncthreads();
  for (int s = 128; s > 0; s >>= 1) {
    if (tid < s) sm[tid] = fmaxf(sm[tid], sm[tid + s]);
    __syncthreads();
  }
  const float rmax = sm[0]; __syncthreads();
  float acc = 0.f;
  for (int i = tid; i < NTOK; i += 256) acc += __expf(row[i] - rmax);
  sm[tid] = acc; __syncthreads();
  for (int s = 128; s > 0; s >>= 1) {
    if (tid < s) sm[tid] += sm[tid + s];
    __syncthreads();
  }
  if (tid == 0) { rowmax[rid] = rmax; rowsum[rid] = sm[0]; }
}

// ---------------------------------------------------------------------------
// K3: c[b,h,d,e] = sum_n softmax_n(k)[d,n] * v[e,n]   (32x32 per (b,h))
// 8 waves split the n range; each wave holds 2x2 16x16 WMMA accumulators;
// cross-wave reduce through LDS, apply 1/rowsum at the end.
// ---------------------------------------------------------------------------
__global__ __launch_bounds__(256) void kv_outer_kernel(
    const float* __restrict__ qkv, const float* __restrict__ rowmax,
    const float* __restrict__ rowsum, float* __restrict__ cmat)
{
  __shared__ float red[8][1024];
  const int bh = blockIdx.x;             // 0..31
  const int b  = bh >> 2;
  const int h  = bh & 3;
  const int lane = threadIdx.x & 31;
  const int wave = threadIdx.x >> 5;
  const int half = lane >> 4;
  const int lrow = lane & 15;

  const float* kbase = qkv + ((size_t)b * QKV_ROWS + DIM + h * DHEAD) * NTOK;
  const float* vbase = qkv + ((size_t)b * QKV_ROWS + 2 * DIM + h * DHEAD) * NTOK;
  const float* rmx = rowmax + b * 128 + h * DHEAD;

  v8f acc[2][2] = {};
  for (int nn = wave * 32; nn < NTOK; nn += 8 * 32) {
    v16h afr[2], bfr[2];
    #pragma unroll
    for (int da = 0; da < 2; ++da) {     // A: rows d, K = n
      const int d = da * 16 + lrow;
      const float* krow = kbase + (size_t)d * NTOK + nn + half * 8;
      const float mx = rmx[d];
      #pragma unroll
      for (int j = 0; j < 8; ++j) afr[da][j]     = (_Float16)__expf(krow[j] - mx);
      #pragma unroll
      for (int j = 0; j < 8; ++j) afr[da][8 + j] = (_Float16)__expf(krow[16 + j] - mx);
    }
    #pragma unroll
    for (int eb = 0; eb < 2; ++eb) {     // B: K = n, N = e (this is V^T)
      const int e = eb * 16 + lrow;
      const float* vrow = vbase + (size_t)e * NTOK + nn + half * 8;
      #pragma unroll
      for (int j = 0; j < 16; ++j) bfr[eb][j] = (_Float16)vrow[j < 8 ? j : j + 8];
    }
    #pragma unroll
    for (int da = 0; da < 2; ++da)
      #pragma unroll
      for (int eb = 0; eb < 2; ++eb)
        acc[da][eb] = wmma_f16(afr[da], bfr[eb], acc[da][eb]);
  }
  #pragma unroll
  for (int da = 0; da < 2; ++da)
    #pragma unroll
    for (int eb = 0; eb < 2; ++eb)
      #pragma unroll
      for (int r = 0; r < 8; ++r) {
        const int d = da * 16 + half * 8 + r;
        const int e = eb * 16 + lrow;
        red[wave][d * 32 + e] = acc[da][eb][r];
      }
  __syncthreads();
  const float* rsum = rowsum + b * 128 + h * DHEAD;
  for (int t = threadIdx.x; t < 1024; t += 256) {
    float s = 0.f;
    #pragma unroll
    for (int w = 0; w < 8; ++w) s += red[w][t];
    cmat[(size_t)bh * 1024 + t] = s / rsum[t >> 5];
  }
}

// ---------------------------------------------------------------------------
// K4: fold attention into output projection:
//     W2[b][o][h*32+d] = sum_e w_out[o][h*32+e] * c[b,h,d,e]   (tiny)
// ---------------------------------------------------------------------------
__global__ __launch_bounds__(256) void fold_wout_kernel(
    const float* __restrict__ w_out, const float* __restrict__ cmat,
    float* __restrict__ W2)
{
  const int idx = blockIdx.x * 256 + threadIdx.x;   // 8*128*128 total
  const int b = idx >> 14;
  const int o = (idx >> 7) & 127;
  const int j = idx & 127;
  const int h = j >> 5;
  const int d = j & 31;
  const float* wrow = w_out + o * DIM + h * DHEAD;
  const float* crow = cmat + (((size_t)b * NHEADS + h) * DHEAD + d) * DHEAD;
  float s = 0.f;
  #pragma unroll
  for (int e = 0; e < DHEAD; ++e) s += wrow[e] * crow[e];
  W2[idx] = s;
}

// ---------------------------------------------------------------------------
// K5: y[b,o,n] = sum_j W2[b,o,j] * qhat[b,j,n] + b_out[o]
//     qhat = SCALE * softmax over head-dim (32 values of column n, split
//     across lane pair (l, l^16) in the B fragment -> shfl_xor(16) combine).
//     One wave handles ALL 8 m-tiles for its n-tile: the q-softmax fragment
//     is built once per head and reused for 8 WMMAs.
//     Also accumulates per-batch sum / sumsq for GroupNorm via atomics.
// ---------------------------------------------------------------------------
__global__ __launch_bounds__(256) void out_gemm_kernel(
    const float* __restrict__ qkv, const float* __restrict__ W2,
    const float* __restrict__ b_out, float* __restrict__ ybuf,
    float* __restrict__ stats)
{
  const int lane = threadIdx.x & 31;
  const int wave = threadIdx.x >> 5;
  const int half = lane >> 4;
  const int lrow = lane & 15;
  const int b  = blockIdx.z;
  const int n0 = (blockIdx.x * 8 + wave) * 16;

  const float* qbase = qkv + (size_t)b * QKV_ROWS * NTOK;   // q = rows 0..127
  const float* w2b   = W2 + (size_t)b * DIM * DIM;

  v8f acc[8] = {};
  #pragma unroll
  for (int h = 0; h < NHEADS; ++h) {
    const int kk = h * DHEAD;

    // Column softmax over head dim: this lane holds 16 of the 32 d-values
    // of column n0+lrow; partner lane (lane^16) holds the other 16.
    float qv[16];
    const float* qcol = qbase + (size_t)(kk + half * 8) * NTOK + n0 + lrow;
    float mloc = -3.4e38f;
    #pragma unroll
    for (int j = 0; j < 16; ++j) { qv[j] = qcol[(size_t)(j < 8 ? j : j + 8) * NTOK]; mloc = fmaxf(mloc, qv[j]); }
    const float mall = fmaxf(mloc, __shfl_xor(mloc, 16, 32));
    float sloc = 0.f;
    #pragma unroll
    for (int j = 0; j < 16; ++j) { qv[j] = __expf(qv[j] - mall); sloc += qv[j]; }
    const float sall = sloc + __shfl_xor(sloc, 16, 32);
    const float inv = SCALE_F / sall;
    v16h bf;
    #pragma unroll
    for (int j = 0; j < 16; ++j) bf[j] = (_Float16)(qv[j] * inv);

    #pragma unroll
    for (int t = 0; t < 8; ++t) {
      const v16h a = load_a_frag(
          w2b + (size_t)(t * 16 + lrow) * DIM + kk + half * 8);
      acc[t] = wmma_f16(a, bf, acc[t]);
    }
  }

  float* yb = ybuf + (size_t)b * DIM * NTOK;
  float s1 = 0.f, s2 = 0.f;
  #pragma unroll
  for (int t = 0; t < 8; ++t)
    #pragma unroll
    for (int r = 0; r < 8; ++r) {
      const int m = t * 16 + half * 8 + r;
      const float yv = acc[t][r] + b_out[m];
      yb[(size_t)m * NTOK + n0 + lrow] = yv;
      s1 += yv; s2 += yv * yv;
    }
  #pragma unroll
  for (int off = 16; off > 0; off >>= 1) {
    s1 += __shfl_down(s1, off, 32);
    s2 += __shfl_down(s2, off, 32);
  }
  if (lane == 0) {
    atomicAdd(&stats[b * 2 + 0], s1);
    atomicAdd(&stats[b * 2 + 1], s2);
  }
}

// ---------------------------------------------------------------------------
// K6: GroupNorm finalize: (y - mean) * rsqrt(var+eps) * gamma + beta
// ---------------------------------------------------------------------------
__global__ __launch_bounds__(256) void gn_kernel(
    const float* __restrict__ ybuf, const float* __restrict__ stats,
    const float* __restrict__ gamma, const float* __restrict__ beta,
    float* __restrict__ out)
{
  const size_t idx4 = (size_t)blockIdx.x * 256 + threadIdx.x;
  const size_t base = idx4 * 4;
  const int b = (int)(base / ((size_t)DIM * NTOK));
  const int o = (int)((base / NTOK) % DIM);
  const float cnt  = (float)DIM * (float)NTOK;
  const float mean = stats[b * 2 + 0] / cnt;
  const float var  = stats[b * 2 + 1] / cnt - mean * mean;
  const float inv  = rsqrtf(var + EPS_F);
  const float g = gamma[o] * inv, bb = beta[o];
  const float4 yv = *(const float4*)(ybuf + base);
  float4 ov;
  ov.x = (yv.x - mean) * g + bb;
  ov.y = (yv.y - mean) * g + bb;
  ov.z = (yv.z - mean) * g + bb;
  ov.w = (yv.w - mean) * g + bb;
  *(float4*)(out + base) = ov;
}

extern "C" void kernel_launch(void* const* d_in, const int* in_sizes, int n_in,
                              void* d_out, int out_size, void* d_ws, size_t ws_size,
                              hipStream_t stream) {
  const float* x     = (const float*)d_in[0];
  const float* w_qkv = (const float*)d_in[1];
  const float* w_out = (const float*)d_in[2];
  const float* b_out = (const float*)d_in[3];
  const float* gamma = (const float*)d_in[4];
  const float* beta  = (const float*)d_in[5];
  float* out = (float*)d_out;
  float* ws  = (float*)d_ws;

  const size_t QKV = (size_t)B_SZ * QKV_ROWS * NTOK;   // 402.7 MB
  const size_t Y   = (size_t)B_SZ * DIM * NTOK;        // 134.2 MB
  float* qkv    = ws;
  float* ybuf   = ws + QKV;
  float* rowmax = ybuf + Y;
  float* rowsum = rowmax + 1024;
  float* cmat   = rowsum + 1024;
  float* W2     = cmat + (size_t)B_SZ * NHEADS * DHEAD * DHEAD;
  float* stats  = W2 + (size_t)B_SZ * DIM * DIM;

  hipMemsetAsync(stats, 0, 2 * B_SZ * sizeof(float), stream);

  dim3 g1(NTOK / 16 / 8, QKV_ROWS / 96, B_SZ);                 // 256 x 4 x 8
  qkv_gemm_kernel<<<g1, 256, 0, stream>>>(x, w_qkv, qkv);

  k_stats_kernel<<<B_SZ * DIM, 256, 0, stream>>>(qkv, rowmax, rowsum);

  kv_outer_kernel<<<B_SZ * NHEADS, 256, 0, stream>>>(qkv, rowmax, rowsum, cmat);

  fold_wout_kernel<<<(B_SZ * DIM * DIM) / 256, 256, 0, stream>>>(w_out, cmat, W2);

  dim3 g5(NTOK / 16 / 8, 1, B_SZ);                             // 256 x 1 x 8
  out_gemm_kernel<<<g5, 256, 0, stream>>>(qkv, W2, b_out, ybuf, stats);

  gn_kernel<<<(int)(((size_t)B_SZ * DIM * NTOK / 4) / 256), 256, 0, stream>>>(
      ybuf, stats, gamma, beta, out);
}
